// ANNAAttention_17609365914146
// MI455X (gfx1250) — matrix operand order repaired and checked by reading, not compiled, not run on hardware
//
#include <hip/hip_runtime.h>
#include <hip/hip_bf16.h>
#include <cstdint>
#include <cstddef>

// ---------------- problem constants (from reference) ----------------
#define NHEADS 8
#define NLM    64          // landmarks / segments
#define TOPK   4
#define NTOK   2048        // N
#define CDIM   512         // C
#define DH     64          // head dim
#define SEG    32          // NTOK / NLM  (exact, no padding)
#define QKVC   1536        // 3*C

typedef __attribute__((ext_vector_type(16))) __bf16 v16bf;
typedef __attribute__((ext_vector_type(8)))  __bf16 v8bf;
typedef __attribute__((ext_vector_type(8)))  float  v8f;

__device__ __forceinline__ unsigned short bf_bits(float f) {
  return __builtin_bit_cast(unsigned short, (__bf16)f);
}

// ---------------------------------------------------------------
// fp32 -> packed bf16 conversion (2 elements / thread)
// ---------------------------------------------------------------
__global__ void cvt_f32_to_bf16(const float* __restrict__ src,
                                __bf16* __restrict__ dst, int npairs) {
  const int i = blockIdx.x * blockDim.x + threadIdx.x;
  if (i >= npairs) return;
  float2 f = ((const float2*)src)[i];
  unsigned pk = (unsigned)bf_bits(f.x) | ((unsigned)bf_bits(f.y) << 16);
  ((unsigned*)dst)[i] = pk;
}

// ---------------------------------------------------------------
// C[M,N] = A[M,K] * Bw[N,K]^T (+ bias); A,Bw bf16 row-major, C fp32.
// One wave32 per 16x64 output tile (4 WMMA accumulators), K-step 32.
// Fragment layouts per CDNA5 ISA 7.12.2 are contiguous in bf16:
//   A 16x32 : lane half hw=lane>>4, row m=lane&15;
//             elems 0..7  = k0+hw*8   .. +7   (one b128)
//             elems 8..15 = k0+16+hw*8.. +7   (one b128)
//   B 32x16 : col n=lane&15; same k layout as A (two b128)
//   C/D     : M = vgpr + 8*hw, N = lane&15
// ---------------------------------------------------------------
__global__ void gemm_nt_wmma(const __bf16* __restrict__ A,
                             const __bf16* __restrict__ Bw,
                             const float* __restrict__ bias,
                             float* __restrict__ C,
                             int M, int N, int K) {
  const int wave = threadIdx.x >> 5;
  const int lane = threadIdx.x & 31;
  const int hw   = lane >> 4;
  const int r    = lane & 15;
  const int tilesN4 = N >> 6;                    // 64-wide N tiles
  const int tile = blockIdx.x * 4 + wave;        // 4 waves / block
  const int tm = (tile / tilesN4) << 4;
  const int tn = (tile % tilesN4) << 6;

  const __bf16* arow = A + (size_t)(tm + r) * K;
  const __bf16* b0   = Bw + (size_t)(tn + r) * K;        // cols tn+r, +16, +32, +48
  const size_t  bstep = (size_t)16 * K;

  v8f acc0 = {}, acc1 = {}, acc2 = {}, acc3 = {};
  for (int k0 = 0; k0 < K; k0 += 32) {
    __builtin_prefetch(arow + k0 + 256, 0, 0);   // next slabs -> global_prefetch_b8
    __builtin_prefetch(b0 + k0 + 256, 0, 0);

    v8bf alo = *(const v8bf*)(arow + k0 + hw * 8);
    v8bf ahi = *(const v8bf*)(arow + k0 + 16 + hw * 8);
    v16bf a = __builtin_shufflevector(alo, ahi, 0,1,2,3,4,5,6,7,8,9,10,11,12,13,14,15);

#pragma unroll
    for (int t = 0; t < 4; ++t) {
      const __bf16* br = b0 + (size_t)t * bstep + k0 + hw * 8;
      v8bf blo = *(const v8bf*)(br);
      v8bf bhi = *(const v8bf*)(br + 16);
      v16bf b = __builtin_shufflevector(blo, bhi, 0,1,2,3,4,5,6,7,8,9,10,11,12,13,14,15);
      v8f* accp = (t == 0) ? &acc0 : (t == 1) ? &acc1 : (t == 2) ? &acc2 : &acc3;
      *accp = __builtin_amdgcn_wmma_f32_16x16x32_bf16(
          false, a, false, b, (short)0, *accp, false, false);
    }
  }

#pragma unroll
  for (int t = 0; t < 4; ++t) {
    const int col = tn + t * 16 + r;
    const float bv = bias ? bias[col] : 0.0f;
    const v8f acc = (t == 0) ? acc0 : (t == 1) ? acc1 : (t == 2) ? acc2 : acc3;
#pragma unroll
    for (int v = 0; v < 8; ++v) {
      const int row = tm + hw * 8 + v;
      C[(size_t)row * N + col] = acc[v] + bv;
    }
  }
}

// ---------------------------------------------------------------
// centroids[h][m][d] = mean over seg of K[h][m*SEG+s][d]
// K lives inside flat qkv: qkv[n*1536 + 512 + h*64 + d]
// ---------------------------------------------------------------
__global__ void centroid_kernel(const float* __restrict__ qkv,
                                float* __restrict__ cent) {
  const int idx = blockIdx.x * blockDim.x + threadIdx.x;
  if (idx >= NHEADS * NLM * DH) return;
  const int d = idx & (DH - 1);
  const int m = (idx >> 6) & (NLM - 1);
  const int h = idx >> 12;
  const float* base = qkv + (size_t)(m * SEG) * QKVC + CDIM + h * DH + d;
  float s = 0.f;
#pragma unroll 4
  for (int t = 0; t < SEG; ++t) s += base[(size_t)t * QKVC];
  cent[idx] = s * (1.0f / SEG);
}

// ---------------------------------------------------------------
// route[h][n][m] = scale * q . centroid ; top-4 argmax (stable, lower
// index wins ties). One wave32 per (h,n); 2 landmarks / lane.
// ---------------------------------------------------------------
__global__ void route_topk_kernel(const float* __restrict__ qkv,
                                  const float* __restrict__ cent,
                                  int* __restrict__ topk) {
  const int wave = threadIdx.x >> 5;
  const int lane = threadIdx.x & 31;
  const int w = blockIdx.x * (blockDim.x >> 5) + wave;   // (h,n) flat
  const int h = w >> 11;
  const int n = w & (NTOK - 1);
  const float scale = 0.125f;      // Dh^-0.5

  const float* q  = qkv + (size_t)n * QKVC + h * DH;     // broadcast reads
  const float* c0 = cent + ((size_t)h * NLM + lane) * DH;
  const float* c1 = c0 + 32 * DH;

  float s0 = 0.f, s1 = 0.f;
#pragma unroll
  for (int d = 0; d < DH; d += 4) {
    float4 qq = *(const float4*)(q + d);
    float4 a  = *(const float4*)(c0 + d);
    float4 b  = *(const float4*)(c1 + d);
    s0 += qq.x * a.x + qq.y * a.y + qq.z * a.z + qq.w * a.w;
    s1 += qq.x * b.x + qq.y * b.y + qq.z * b.z + qq.w * b.w;
  }
  s0 *= scale; s1 *= scale;

  int* out = topk + ((size_t)h * NTOK + n) * TOPK;
#pragma unroll
  for (int t = 0; t < TOPK; ++t) {
    float lv; int li;
    if (s0 >= s1) { lv = s0; li = lane; } else { lv = s1; li = lane + 32; }
#pragma unroll
    for (int off = 16; off > 0; off >>= 1) {             // wave32 argmax reduce
      float ov = __shfl_xor(lv, off, 32);
      int   oi = __shfl_xor(li, off, 32);
      if (ov > lv || (ov == lv && oi < li)) { lv = ov; li = oi; }
    }
    if (lane == 0) out[t] = li;
    if (li == lane)      s0 = -3.4e38f;                  // remove winner
    if (li == lane + 32) s1 = -3.4e38f;
  }
}

// ---------------------------------------------------------------
// Gathered attention: per query, 4 selected segments x 32 keys = 128 kv.
// One wave32 per (h,n). Lane s scores key s of each segment; softmax via
// wave reductions; output dims split 2 per lane; result stored as packed
// bf16 (feeds the bf16-WMMA output projection).
// ---------------------------------------------------------------
__global__ void attn_kernel(const float* __restrict__ qkv,
                            const int* __restrict__ topk,
                            __bf16* __restrict__ attn_bf) {
  __shared__ __align__(16) float qs[8][DH];
  __shared__ __align__(16) float ps[8][TOPK * SEG];

  const int wave = threadIdx.x >> 5;
  const int lane = threadIdx.x & 31;
  const int w = blockIdx.x * 8 + wave;
  const int h = w >> 11;
  const int n = w & (NTOK - 1);
  const float scale = 0.125f;

  // stage q row into LDS (2 floats / lane)
  const float* q = qkv + (size_t)n * QKVC + h * DH;
  float2 qv = *(const float2*)(q + lane * 2);
  qs[wave][lane * 2]     = qv.x;
  qs[wave][lane * 2 + 1] = qv.y;

  const int* tkp = topk + ((size_t)h * NTOK + n) * TOPK;
  int sg[TOPK];
#pragma unroll
  for (int u = 0; u < TOPK; ++u) sg[u] = tkp[u];

  __syncthreads();

  // scores: lane handles key index `lane` of each of the 4 segments
  const float4* qp = (const float4*)qs[wave];
  float sc[TOPK];
#pragma unroll
  for (int u = 0; u < TOPK; ++u) {
    const int row = sg[u] * SEG + lane;
    const float* kr = qkv + (size_t)row * QKVC + CDIM + h * DH;
    float s = 0.f;
#pragma unroll
    for (int d = 0; d < DH; d += 4) {
      float4 qq = qp[d >> 2];
      float4 kk = *(const float4*)(kr + d);
      s += qq.x * kk.x + qq.y * kk.y + qq.z * kk.z + qq.w * kk.w;
    }
    sc[u] = s * scale;
  }

  // softmax over 128 (4 per lane x 32 lanes)
  float mx = sc[0];
#pragma unroll
  for (int u = 1; u < TOPK; ++u) mx = fmaxf(mx, sc[u]);
#pragma unroll
  for (int off = 16; off > 0; off >>= 1) mx = fmaxf(mx, __shfl_xor(mx, off, 32));
  float e[TOPK], sum = 0.f;
#pragma unroll
  for (int u = 0; u < TOPK; ++u) { e[u] = __expf(sc[u] - mx); sum += e[u]; }
#pragma unroll
  for (int off = 16; off > 0; off >>= 1) sum += __shfl_xor(sum, off, 32);
  const float inv = 1.0f / sum;
#pragma unroll
  for (int u = 0; u < TOPK; ++u) ps[wave][u * SEG + lane] = e[u] * inv;

  __syncthreads();

  // output: lane owns dims d0 = 2*lane, 2*lane+1
  const int d0 = lane * 2;
  float a0 = 0.f, a1 = 0.f;
#pragma unroll
  for (int u = 0; u < TOPK; ++u) {
    const int base = sg[u] * SEG;
    const float* vb = qkv + (size_t)base * QKVC + 2 * CDIM + h * DH + d0;
    for (int s = 0; s < SEG; ++s) {
      const float wgt = ps[wave][u * SEG + s];
      const float* vr = vb + (size_t)s * QKVC;
      a0 += wgt * vr[0];
      a1 += wgt * vr[1];
    }
  }
  const unsigned pk = (unsigned)bf_bits(a0) | ((unsigned)bf_bits(a1) << 16);
  *(unsigned*)(attn_bf + (size_t)n * CDIM + h * DH + d0) = pk;
}

// ---------------------------------------------------------------
extern "C" void kernel_launch(void* const* d_in, const int* in_sizes, int n_in,
                              void* d_out, int out_size, void* d_ws, size_t ws_size,
                              hipStream_t stream) {
  const float* x      = (const float*)d_in[0];   // (1,2048,512)
  const float* w_qkv  = (const float*)d_in[1];   // (1536,512)
  const float* w_proj = (const float*)d_in[2];   // (512,512)
  const float* b_proj = (const float*)d_in[3];   // (512,)
  float* out = (float*)d_out;                    // (1,2048,512)

  // ---- workspace layout (offsets stay 256B-aligned) ----
  char* ws = (char*)d_ws;
  size_t off = 0;
  float* qkv  = (float*)(ws + off); off += (size_t)NTOK * QKVC * sizeof(float);
  float* cent = (float*)(ws + off); off += (size_t)NHEADS * NLM * DH * sizeof(float);
  int*   tk   = (int*)  (ws + off); off += (size_t)NHEADS * NTOK * TOPK * sizeof(int);
  __bf16* x_bf  = (__bf16*)(ws + off); off += (size_t)NTOK * CDIM * sizeof(__bf16);
  __bf16* wq_bf = (__bf16*)(ws + off); off += (size_t)QKVC * CDIM * sizeof(__bf16);
  __bf16* wp_bf = (__bf16*)(ws + off); off += (size_t)CDIM * CDIM * sizeof(__bf16);
  __bf16* at_bf = (__bf16*)(ws + off); off += (size_t)NTOK * CDIM * sizeof(__bf16);

  // 0) one-shot bf16 conversion of GEMM operands
  cvt_f32_to_bf16<<<(NTOK * CDIM / 2) / 256, 256, 0, stream>>>(x, x_bf, NTOK * CDIM / 2);
  cvt_f32_to_bf16<<<(QKVC * CDIM / 2) / 256, 256, 0, stream>>>(w_qkv, wq_bf, QKVC * CDIM / 2);
  cvt_f32_to_bf16<<<(CDIM * CDIM / 2) / 256, 256, 0, stream>>>(w_proj, wp_bf, CDIM * CDIM / 2);

  // 1) QKV projection: (2048x512) x (512x1536), 16x64 tile / wave
  gemm_nt_wmma<<<(NTOK / 16) * (QKVC / 64) / 4, 128, 0, stream>>>(
      x_bf, wq_bf, nullptr, qkv, NTOK, QKVC, CDIM);

  // 2) segment centroids of K
  centroid_kernel<<<(NHEADS * NLM * DH) / 256, 256, 0, stream>>>(qkv, cent);

  // 3) routing scores + top-4 segment selection (wave32 per query)
  route_topk_kernel<<<NHEADS * NTOK / 8, 256, 0, stream>>>(qkv, cent, tk);

  // 4) gathered attention over 4x32 keys per query -> bf16 output
  attn_kernel<<<NHEADS * NTOK / 8, 256, 0, stream>>>(qkv, tk, at_bf);

  // 5) output projection + bias via bf16 WMMA -> fp32 d_out
  gemm_nt_wmma<<<(NTOK / 16) * (CDIM / 64) / 4, 128, 0, stream>>>(
      at_bf, wp_bf, b_proj, out, NTOK, CDIM, CDIM);
}
